// SkeletonGNN_30760555774065
// MI455X (gfx1250) — matrix operand order repaired and checked by reading, not compiled, not run on hardware
//
#include <hip/hip_runtime.h>

typedef __attribute__((ext_vector_type(16))) _Float16 v16h;
typedef __attribute__((ext_vector_type(8)))  _Float16 v8h;
typedef __attribute__((ext_vector_type(8)))  float    v8f;

#define N_NODES 26000
#define N_EDGES 208000
#define N_GRAPH 1040
#define FRAMES  65
#define F_INF   6
#define HID     128
#define HEADS   4
#define NCLS    7

// ---------- small helpers ----------
__device__ __forceinline__ void atomic_max_f32(float* addr, float v) {
    if (v >= 0.0f) atomicMax((int*)addr, __float_as_int(v));
    else           atomicMin((unsigned int*)addr, __float_as_uint(v));
}

// CDNA5 async copy: global -> LDS, 16B per lane, tracked by ASYNCcnt.
__device__ __forceinline__ void async_b128_to_lds(unsigned lds_off, const void* gptr) {
    unsigned long long ga = (unsigned long long)gptr;
    asm volatile("global_load_async_to_lds_b128 %0, %1, off"
                 :: "v"(lds_off), "v"(ga) : "memory");
}
__device__ __forceinline__ void wait_asynccnt0() {
    asm volatile("s_wait_asynccnt 0x0" ::: "memory");
}

__global__ void fill_f32(float* __restrict__ p, float v, long long n) {
    long long i = (long long)blockIdx.x * blockDim.x + threadIdx.x;
    if (i < n) p[i] = v;
}

// ---------- stage 0: GraphConv1 (K=6, scalar path) ----------
__global__ void scatter_x6(const float* __restrict__ x, const int* __restrict__ src,
                           const int* __restrict__ dst, float* __restrict__ agg6) {
    long long e = (long long)blockIdx.x * blockDim.x + threadIdx.x;
    if (e >= N_EDGES) return;
    int s = src[e], d = dst[e];
#pragma unroll
    for (int f = 0; f < F_INF; ++f)
        atomicAdd(&agg6[(long long)d * F_INF + f], x[(long long)s * F_INF + f]);
}

__global__ void gconv1(const float* __restrict__ x, const float* __restrict__ agg6,
                       const float* __restrict__ w1r, const float* __restrict__ w1n,
                       const float* __restrict__ b1,
                       float* __restrict__ h1, _Float16* __restrict__ h1h) {
    int n = blockIdx.x, c = threadIdx.x;
    __shared__ float xs[F_INF], as[F_INF];
    if (c < F_INF) { xs[c] = x[(long long)n * F_INF + c]; as[c] = agg6[(long long)n * F_INF + c]; }
    __syncthreads();
    float v = b1[c];
#pragma unroll
    for (int f = 0; f < F_INF; ++f)
        v += xs[f] * w1r[f * HID + c] + as[f] * w1n[f * HID + c];
    v = fmaxf(v, 0.0f);
    h1[(long long)n * HID + c] = v;
    h1h[(long long)n * HID + c] = (_Float16)v;
}

// ---------- weight pack into WMMA B-fragment order ----------
// B (32x16 f16): lane holds column n = lane&15; element s holds K = 8*(lane>>4) + (s&7) + 16*(s>>3)
// packed halves: [ ((nt*ksteps+ks)*32 + lane)*16 + i*2 + h ]
__global__ void pack_w(const float* __restrict__ W0, const float* __restrict__ W1,
                       int mode, int K0, int Nout, int ksteps, int ntiles,
                       _Float16* __restrict__ out) {
    long long idx = (long long)blockIdx.x * blockDim.x + threadIdx.x;
    long long tot = (long long)ntiles * ksteps * 512;
    if (idx >= tot) return;
    int e    = (int)(idx & 511);
    int lane = e >> 4;
    int sub  = e & 15;           // i*2+h
    long long t = idx >> 9;      // nt*ksteps + ks
    int ks = (int)(t % ksteps);
    int nt = (int)(t / ksteps);
    int n = nt * 16 + (lane & 15);
    int k = ks * 32 + ((lane >> 4) << 3) + (sub & 7) + ((sub >> 3) << 4);
    float v;
    if (mode == 0)      v = W0[(long long)k * Nout + n];
    else if (mode == 1) v = (k < K0) ? W0[(long long)k * Nout + n]
                                     : W1[(long long)(k - K0) * Nout + n];
    else                v = W0[(long long)k * Nout + n] + W1[(long long)k * Nout + n];
    out[idx] = (_Float16)v;
}

// ---------- WMMA GEMM: each wave computes a 1x4 strip of 16x16 tiles ----------
// Block = 4 waves (4 m-tiles) sharing one group of 4 n-tiles whose packed B is
// staged into LDS with global_load_async_to_lds_b128 (ASYNCcnt) once per block.
__global__ void __launch_bounds__(128)
gemm_wmma_f16(const _Float16* __restrict__ A, int lda,
              const _Float16* __restrict__ Wp,
              const float* __restrict__ bias,
              const float* __restrict__ resid,
              float* __restrict__ outF,
              _Float16* __restrict__ outH, int ldh,
              int ldc, int mtiles, int ksteps, int do_relu) {
    extern __shared__ _Float16 bsh[];           // 4 * ksteps * 512 halves
    const int ng   = blockIdx.x;                // group of 4 n-tiles
    const int lane = threadIdx.x;
    const int tid  = threadIdx.y * 32 + lane;

    // ---- stage packed B (contiguous, nt-major) into LDS, all 4 waves help ----
    {
        const char* wsrc = (const char*)(Wp + (size_t)ng * 4 * ksteps * 512);
        unsigned lbase = (unsigned)(uintptr_t)bsh;
        int nchunk = ksteps * 256;              // 16B chunks (4*ksteps*1024B / 16)
        for (int c = tid; c < nchunk; c += 128)
            async_b128_to_lds(lbase + c * 16, wsrc + (size_t)c * 16);
        wait_asynccnt0();
        __syncthreads();
    }

    const int mtile = blockIdx.y * blockDim.y + threadIdx.y;
    if (mtile >= mtiles) return;                // wave-uniform exit (after barrier)
    const int r = lane & 15, hi = lane >> 4;

    // A fragment (16x32 f16): lane r(+16*hi) -> row r; halves 8*hi..+7, 16+8*hi..+7
    const _Float16* arow = A + (long long)(mtile * 16 + r) * lda + 8 * hi;

    v8f acc0 = {0.f,0.f,0.f,0.f,0.f,0.f,0.f,0.f};
    v8f acc1 = acc0, acc2 = acc0, acc3 = acc0;

    for (int ks = 0; ks < ksteps; ++ks) {
        v8h a0 = *(const v8h*)(arow + ks * 32);
        v8h a1 = *(const v8h*)(arow + ks * 32 + 16);
        v16h av;
#pragma unroll
        for (int i = 0; i < 8; ++i) { av[i] = a0[i]; av[i + 8] = a1[i]; }
        const _Float16* bks = bsh + ((size_t)ks * 32 + lane) * 16;
        v16h b0 = *(const v16h*)(bks);
        v16h b1v = *(const v16h*)(bks + (size_t)ksteps * 512 * 1);
        v16h b2 = *(const v16h*)(bks + (size_t)ksteps * 512 * 2);
        v16h b3 = *(const v16h*)(bks + (size_t)ksteps * 512 * 3);
        acc0 = __builtin_amdgcn_wmma_f32_16x16x32_f16(false, av, false, b0, (short)0, acc0, false, false);
        acc1 = __builtin_amdgcn_wmma_f32_16x16x32_f16(false, av, false, b1v, (short)0, acc1, false, false);
        acc2 = __builtin_amdgcn_wmma_f32_16x16x32_f16(false, av, false, b2, (short)0, acc2, false, false);
        acc3 = __builtin_amdgcn_wmma_f32_16x16x32_f16(false, av, false, b3, (short)0, acc3, false, false);
    }

    v8f accs[4] = {acc0, acc1, acc2, acc3};
#pragma unroll
    for (int t = 0; t < 4; ++t) {
        int col = (ng * 4 + t) * 16 + r;
        float b = bias ? bias[col] : 0.0f;
#pragma unroll
        for (int i = 0; i < 8; ++i) {
            int m = mtile * 16 + i + 8 * hi;    // C/D: VGPR i, lanes16-31 -> M=i+8
            float v = accs[t][i] + b;
            if (resid) v += resid[(long long)m * ldc + col];
            if (do_relu) v = fmaxf(v, 0.0f);
            if (outF) outF[(long long)m * ldc + col] = v;
            if (outH) outH[(long long)m * ldh + col] = (_Float16)v;
        }
    }
}

// ---------- attention ----------
__global__ void attn_logits(const _Float16* __restrict__ qkvh,
                            const int* __restrict__ src, const int* __restrict__ dst,
                            float* __restrict__ logits, float* __restrict__ mbuf) {
    long long idx = (long long)blockIdx.x * blockDim.x + threadIdx.x;
    if (idx >= (long long)N_EDGES * HEADS) return;
    int e = (int)(idx >> 2), hh = (int)(idx & 3);
    int s = src[e], d = dst[e];
    const v8h* q8 = (const v8h*)(qkvh + (long long)d * 1536 + hh * HID);        // q
    const v8h* k8 = (const v8h*)(qkvh + (long long)s * 1536 + 512 + hh * HID);  // k
    float acc = 0.f;
#pragma unroll
    for (int i = 0; i < 16; ++i) {
        v8h a = q8[i], b = k8[i];
#pragma unroll
        for (int j = 0; j < 8; ++j) acc += (float)a[j] * (float)b[j];
    }
    float lg = acc * 0.08838834764831845f;      // 1/sqrt(128)
    logits[idx] = lg;
    atomic_max_f32(&mbuf[(long long)d * HEADS + hh], lg);
}

__global__ void attn_denom(const int* __restrict__ dst,
                           float* __restrict__ logits,
                           const float* __restrict__ mbuf, float* __restrict__ denom) {
    long long idx = (long long)blockIdx.x * blockDim.x + threadIdx.x;
    if (idx >= (long long)N_EDGES * HEADS) return;
    int e = (int)(idx >> 2), hh = (int)(idx & 3);
    int d = dst[e];
    float a = __expf(logits[idx] - mbuf[(long long)d * HEADS + hh]);
    logits[idx] = a;
    atomicAdd(&denom[(long long)d * HEADS + hh], a);
}

__global__ void attn_norm(const int* __restrict__ dst, float* __restrict__ logits,
                          const float* __restrict__ denom) {
    long long idx = (long long)blockIdx.x * blockDim.x + threadIdx.x;
    if (idx >= (long long)N_EDGES * HEADS) return;
    int e = (int)(idx >> 2), hh = (int)(idx & 3);
    logits[idx] = logits[idx] / denom[(long long)dst[e] * HEADS + hh];
}

__global__ void attn_scatter_v(const _Float16* __restrict__ qkvh,
                               const int* __restrict__ src, const int* __restrict__ dst,
                               const float* __restrict__ logits,
                               float* __restrict__ big0) {
    long long idx = (long long)blockIdx.x * blockDim.x + threadIdx.x;
    if (idx >= (long long)N_EDGES * 512) return;
    int e = (int)(idx >> 9), j = (int)(idx & 511), hh = j >> 7;
    int d = dst[e], s = src[e];
    float w = logits[(long long)e * HEADS + hh];
    atomicAdd(&big0[(long long)d * 512 + j],
              w * (float)qkvh[(long long)s * 1536 + 1024 + j]);     // v
}

__global__ void relu_to_f16(const float* __restrict__ in, _Float16* __restrict__ out,
                            long long n) {
    long long i = (long long)blockIdx.x * blockDim.x + threadIdx.x;
    if (i < n) out[i] = (_Float16)fmaxf(in[i], 0.0f);
}

// ---------- residual GraphConv helpers ----------
__global__ void scatter128(const float* __restrict__ h, const int* __restrict__ src,
                           const int* __restrict__ dst, float* __restrict__ agg) {
    long long idx = (long long)blockIdx.x * blockDim.x + threadIdx.x;
    if (idx >= (long long)N_EDGES * HID) return;
    int e = (int)(idx >> 7), c = (int)(idx & 127);
    atomicAdd(&agg[(long long)dst[e] * HID + c], h[(long long)src[e] * HID + c]);
}

__global__ void cvt_to_cat(const float* __restrict__ in, _Float16* __restrict__ out,
                           int coloff) {
    long long idx = (long long)blockIdx.x * blockDim.x + threadIdx.x;
    if (idx >= (long long)N_NODES * HID) return;
    int n = (int)(idx >> 7), c = (int)(idx & 127);
    out[(long long)n * 256 + coloff + c] = (_Float16)in[idx];
}

// ---------- pooling + classifier ----------
__global__ void pool_graph(const float* __restrict__ h, float* __restrict__ pooled) {
    int g = blockIdx.x, c = threadIdx.x;
    float s = 0.f;
#pragma unroll
    for (int i = 0; i < 25; ++i) s += h[(long long)(g * 25 + i) * HID + c];
    pooled[(long long)g * HID + c] = s * (1.0f / 25.0f);
}

__global__ void frames_fc(const float* __restrict__ pooled,
                          const float* __restrict__ fcw, const float* __restrict__ fcb,
                          float* __restrict__ out) {
    int s = blockIdx.x, c = threadIdx.x;
    __shared__ float zs[HID];
    float acc = 0.f;
    for (int f = 0; f < FRAMES; ++f) acc += pooled[(long long)(s * FRAMES + f) * HID + c];
    zs[c] = acc * (1.0f / (float)FRAMES);
    __syncthreads();
    if (c < NCLS) {
        float o = fcb[c];
#pragma unroll 8
        for (int k = 0; k < HID; ++k) o += zs[k] * fcw[k * NCLS + c];
        out[s * NCLS + c] = o;
    }
}

// =====================================================================
extern "C" void kernel_launch(void* const* d_in, const int* in_sizes, int n_in,
                              void* d_out, int out_size, void* d_ws, size_t ws_size,
                              hipStream_t stream) {
    (void)in_sizes; (void)n_in; (void)out_size; (void)ws_size;
    const float* x     = (const float*)d_in[0];
    const int*   src   = (const int*)d_in[1];
    const int*   dst   = src + N_EDGES;
    const float* w1r   = (const float*)d_in[3];
    const float* w1n   = (const float*)d_in[4];
    const float* b1    = (const float*)d_in[5];
    const float* wq    = (const float*)d_in[6];
    const float* bq    = (const float*)d_in[7];
    const float* wk    = (const float*)d_in[8];
    const float* bk    = (const float*)d_in[9];
    const float* wv    = (const float*)d_in[10];
    const float* bv    = (const float*)d_in[11];
    const float* wskip = (const float*)d_in[12];
    const float* bskip = (const float*)d_in[13];
    const float* skipw = (const float*)d_in[14];
    const float* l1w   = (const float*)d_in[15];
    const float* l1b   = (const float*)d_in[16];
    const float* l2r   = (const float*)d_in[17];
    const float* l2n   = (const float*)d_in[18];
    const float* l2b   = (const float*)d_in[19];
    const float* l3r   = (const float*)d_in[20];
    const float* l3n   = (const float*)d_in[21];
    const float* l3b   = (const float*)d_in[22];
    const float* fcw   = (const float*)d_in[23];
    const float* fcb   = (const float*)d_in[24];
    float* out = (float*)d_out;

    // ---- workspace carve ----
    char* wsp = (char*)d_ws;
    auto carve = [&](size_t bytes) -> void* {
        void* p = (void*)wsp; wsp += (bytes + 255) & ~(size_t)255; return p;
    };
    float*     agg6  = (float*)carve((size_t)N_NODES * F_INF * 4);
    float*     h1    = (float*)carve((size_t)N_NODES * HID * 4);        // also h3/h5
    _Float16*  h1h   = (_Float16*)carve((size_t)N_NODES * HID * 2);
    _Float16*  qkvh  = (_Float16*)carve((size_t)N_NODES * 1536 * 2);    // reused as h2h
    float*     big0  = (float*)carve((size_t)N_NODES * 512 * 4);
    float*     logits= (float*)carve((size_t)N_EDGES * HEADS * 4);
    float*     mbuf  = (float*)carve((size_t)N_NODES * HEADS * 4);
    float*     denom = (float*)carve((size_t)N_NODES * HEADS * 4);
    float*     aggN  = (float*)carve((size_t)N_NODES * HID * 4);
    _Float16*  catA  = (_Float16*)carve((size_t)N_NODES * 256 * 2);
    _Float16*  catB  = (_Float16*)carve((size_t)N_NODES * 256 * 2);
    float*     bufB  = (float*)carve((size_t)N_NODES * HID * 4);        // h4
    float*     pooled= (float*)carve((size_t)N_GRAPH * HID * 4);
    _Float16*  qkvwp = (_Float16*)carve((size_t)96 * 4 * 512 * 2);
    _Float16*  skipp = (_Float16*)carve((size_t)32 * 4 * 512 * 2);
    _Float16*  l1wp  = (_Float16*)carve((size_t)8 * 16 * 512 * 2);
    _Float16*  l2wp  = (_Float16*)carve((size_t)8 * 8 * 512 * 2);
    _Float16*  l3wp  = (_Float16*)carve((size_t)8 * 8 * 512 * 2);
    float*     biasc = (float*)carve(1536 * 4);

    const int TPB = 256;
    const int mtilesN = N_NODES / 16;       // 1625
    dim3 gblk(32, 4);
    unsigned gy = (unsigned)((mtilesN + 3) / 4);
    auto blocks = [](long long n) { return (unsigned)((n + 255) / 256); };

    // ---- init accumulators ----
    fill_f32<<<blocks((long long)N_NODES * F_INF), TPB, 0, stream>>>(agg6, 0.f, (long long)N_NODES * F_INF);
    fill_f32<<<blocks((long long)N_NODES * HEADS), TPB, 0, stream>>>(mbuf, -INFINITY, (long long)N_NODES * HEADS);
    fill_f32<<<blocks((long long)N_NODES * HEADS), TPB, 0, stream>>>(denom, 0.f, (long long)N_NODES * HEADS);

    // ---- pack weights (bias concat via d2d copies) ----
    long long pq = (long long)32 * 4 * 512;
    pack_w<<<blocks(pq), TPB, 0, stream>>>(wq, nullptr, 0, 0, 512, 4, 32, qkvwp);
    pack_w<<<blocks(pq), TPB, 0, stream>>>(wk, nullptr, 0, 0, 512, 4, 32, qkvwp + pq);
    pack_w<<<blocks(pq), TPB, 0, stream>>>(wv, nullptr, 0, 0, 512, 4, 32, qkvwp + 2 * pq);
    pack_w<<<blocks(pq), TPB, 0, stream>>>(wskip, skipw, 2, 0, 512, 4, 32, skipp);
    pack_w<<<blocks((long long)8 * 16 * 512), TPB, 0, stream>>>(l1w, nullptr, 0, 0, HID, 16, 8, l1wp);
    pack_w<<<blocks((long long)8 * 8 * 512), TPB, 0, stream>>>(l2r, l2n, 1, HID, HID, 8, 8, l2wp);
    pack_w<<<blocks((long long)8 * 8 * 512), TPB, 0, stream>>>(l3r, l3n, 1, HID, HID, 8, 8, l3wp);
    hipMemcpyAsync(biasc,        bq, 512 * 4, hipMemcpyDeviceToDevice, stream);
    hipMemcpyAsync(biasc + 512,  bk, 512 * 4, hipMemcpyDeviceToDevice, stream);
    hipMemcpyAsync(biasc + 1024, bv, 512 * 4, hipMemcpyDeviceToDevice, stream);

    // ---- stage 1: GraphConv1 ----
    scatter_x6<<<blocks(N_EDGES), TPB, 0, stream>>>(x, src, dst, agg6);
    gconv1<<<N_NODES, HID, 0, stream>>>(x, agg6, w1r, w1n, b1, h1, h1h);

    // ---- stage 2: q/k/v GEMM + fused (wskip+skip_w) GEMM (WMMA + async-LDS) ----
    gemm_wmma_f16<<<dim3(24, gy), gblk, 4 * 4 * 4096, stream>>>(
        h1h, HID, qkvwp, biasc, nullptr, nullptr, qkvh, 1536, 1536, mtilesN, 4, 0);
    gemm_wmma_f16<<<dim3(8, gy), gblk, 4 * 4 * 4096 / 4, stream>>>(
        h1h, HID, skipp, bskip, nullptr, big0, nullptr, 0, 512, mtilesN, 4, 0);

    // ---- attention (softmax over incoming edges per head) ----
    attn_logits<<<blocks((long long)N_EDGES * HEADS), TPB, 0, stream>>>(qkvh, src, dst, logits, mbuf);
    attn_denom<<<blocks((long long)N_EDGES * HEADS), TPB, 0, stream>>>(dst, logits, mbuf, denom);
    attn_norm<<<blocks((long long)N_EDGES * HEADS), TPB, 0, stream>>>(dst, logits, denom);
    attn_scatter_v<<<blocks((long long)N_EDGES * 512), TPB, 0, stream>>>(qkvh, src, dst, logits, big0);

    // h2 = relu(big0) -> f16 (reuse qkvh storage, dense [N,512])
    _Float16* h2h = qkvh;
    relu_to_f16<<<blocks((long long)N_NODES * 512), TPB, 0, stream>>>(big0, h2h, (long long)N_NODES * 512);

    // ---- lin1: [N,512]x[512,128], relu; f32 -> h1(buf), f16 -> catA[:,0:128] ----
    gemm_wmma_f16<<<dim3(2, gy), gblk, 16 * 4096, stream>>>(
        h2h, 512, l1wp, l1b, nullptr, h1, catA, 256, HID, mtilesN, 16, 1);

    // ---- l2: agg + concat-K GEMM + residual ----
    fill_f32<<<blocks((long long)N_NODES * HID), TPB, 0, stream>>>(aggN, 0.f, (long long)N_NODES * HID);
    scatter128<<<blocks((long long)N_EDGES * HID), TPB, 0, stream>>>(h1, src, dst, aggN);
    cvt_to_cat<<<blocks((long long)N_NODES * HID), TPB, 0, stream>>>(aggN, catA, HID);
    gemm_wmma_f16<<<dim3(2, gy), gblk, 8 * 4096, stream>>>(
        catA, 256, l2wp, l2b, h1, bufB, catB, 256, HID, mtilesN, 8, 1);

    // ---- l3 ----
    fill_f32<<<blocks((long long)N_NODES * HID), TPB, 0, stream>>>(aggN, 0.f, (long long)N_NODES * HID);
    scatter128<<<blocks((long long)N_EDGES * HID), TPB, 0, stream>>>(bufB, src, dst, aggN);
    cvt_to_cat<<<blocks((long long)N_NODES * HID), TPB, 0, stream>>>(aggN, catB, HID);
    gemm_wmma_f16<<<dim3(2, gy), gblk, 8 * 4096, stream>>>(
        catB, 256, l3wp, l3b, bufB, h1, nullptr, 0, HID, mtilesN, 8, 1);

    // ---- pooling + frame average + classifier ----
    pool_graph<<<N_GRAPH, HID, 0, stream>>>(h1, pooled);
    frames_fc<<<N_GRAPH / FRAMES, HID, 0, stream>>>(pooled, fcw, fcb, out);
}
